// DiPaFBackbone_60318520705727
// MI455X (gfx1250) — matrix operand order repaired and checked by reading, not compiled, not run on hardware
//
#include <hip/hip_runtime.h>

#define DEV_INLINE __device__ __forceinline__

typedef __attribute__((ext_vector_type(16))) __bf16 bf16x16;
typedef __attribute__((ext_vector_type(8)))  float  floatx8;

union FragAB { bf16x16 v; unsigned int u[8]; };

DEV_INLINE unsigned short f2bf(float f) {
    union { float f; unsigned u; } c; c.f = f;
    unsigned r = c.u + 0x7FFFu + ((c.u >> 16) & 1u);
    return (unsigned short)(r >> 16);
}

// A fragment: 16-bit A 16x32 (MxK). lane: m=lane&15, g=lane>>4.
// elem e -> k = 16*g + e (contiguous in A row) -> 8 dword loads.
DEV_INLINE bf16x16 load_frag_a(const unsigned short* A, int ld, int lane) {
    int m = lane & 15, g = lane >> 4;
    const unsigned short* row = A + m * ld;
    const unsigned int* p0 = (const unsigned int*)(row + 16 * g);
    FragAB f;
#pragma unroll
    for (int e = 0; e < 8; ++e) { f.u[e] = p0[e]; }
    return f.v;
}

// B fragment from Bt[n][k] (pre-transposed weights): lane n=lane&15, g=lane>>4,
// elem e -> k = 16*g + e (contiguous in Bt row) -> 8 dword loads.
DEV_INLINE bf16x16 load_frag_b(const unsigned short* Bt, int ld, int lane) {
    int n = lane & 15, g = lane >> 4;
    const unsigned int* p = (const unsigned int*)(Bt + n * ld + 16 * g);
    FragAB f;
#pragma unroll
    for (int e = 0; e < 8; ++e) f.u[e] = p[e];
    return f.v;
}

DEV_INLINE floatx8 wmma_bf16(bf16x16 a, bf16x16 b, floatx8 c) {
    return __builtin_amdgcn_wmma_f32_16x16x32_bf16(false, a, false, b,
                                                   (short)0, c, false, false);
}

DEV_INLINE floatx8 zero8() {
    floatx8 z;
#pragma unroll
    for (int e = 0; e < 8; ++e) z[e] = 0.f;
    return z;
}

DEV_INLINE float wave_sum(float s) {
    for (int o = 16; o; o >>= 1) s += __shfl_xor(s, o, 32);
    return s;
}

// ---------------- prep kernels ----------------
__global__ void k_transpose_bf16(const float* __restrict__ src,
                                 unsigned short* __restrict__ dst, int R, int C) {
    int i = blockIdx.x * blockDim.x + threadIdx.x;
    if (i < R * C) { int r = i / C, c = i % C; dst[c * R + r] = f2bf(src[i]); }
}

__global__ void k_convert_bf16(const float* __restrict__ src,
                               unsigned short* __restrict__ dst, int nelem) {
    int i = blockIdx.x * blockDim.x + threadIdx.x;
    if (i < nelem) dst[i] = f2bf(src[i]);
}

// row-wise sum of squares over 128 columns; one wave per row
__global__ __launch_bounds__(256) void k_rowsq(const float* __restrict__ src,
                                               float* __restrict__ dst, int nrows) {
    int r = blockIdx.x * 8 + (threadIdx.x >> 5);
    int lane = threadIdx.x & 31;
    if (r >= nrows) return;
    const float* row = src + (size_t)r * 128;
    float s = 0.f;
#pragma unroll
    for (int j = 0; j < 4; ++j) { float v = row[lane * 4 + j]; s += v * v; }
    s = wave_sum(s);
    if (lane == 0) dst[r] = s;
}

// ---------------- RevIN stats: per (b,v) mean/std over L=1024 ----------------
__global__ __launch_bounds__(256) void k_revin(const float* __restrict__ x,
                                               float* __restrict__ meanA,
                                               float* __restrict__ stdA,
                                               float* __restrict__ rstdA) {
    __shared__ float rs[256], rq[256];
    int bv = blockIdx.x, b = bv >> 6, v = bv & 63, t = threadIdx.x;
    float s = 0.f, q = 0.f;
    for (int i = t; i < 1024; i += 256) {
        float xv = x[((size_t)b * 1024 + i) * 64 + v];
        s += xv; q += xv * xv;
    }
    rs[t] = s; rq[t] = q;
    __syncthreads();
    for (int o = 128; o; o >>= 1) {
        if (t < o) { rs[t] += rs[t + o]; rq[t] += rq[t + o]; }
        __syncthreads();
    }
    if (t == 0) {
        float m = rs[0] * (1.f / 1024.f);
        float var = rq[0] * (1.f / 1024.f) - m * m;
        float sd = sqrtf(var + 1e-5f);
        meanA[bv] = m; stdA[bv] = sd; rstdA[bv] = 1.f / sd;
    }
}

// ---------------- fused encode + LN + temporal MLP (WMMA) ----------------
// One block per (b, v, d-half). LDS (64KB) time-multiplexed:
//   A0   bf16[64][64]   @ 0      (LN'd, transposed: rows = d_local, cols = ph)
//   xn   f32 [1024]     @ 8192
//   zbuf f32 [64][128]  @ 32768
//   h1   bf16[64][256]  @ 32768  (after zbuf dead)
//   h2   bf16[64][256]  @ 0      (after A0/xn dead)
__global__ __launch_bounds__(256) void k_mlp(
    const float* __restrict__ x,
    const float* __restrict__ enc_w, const float* __restrict__ enc_b,
    const float* __restrict__ ln_w, const float* __restrict__ ln_b,
    const unsigned short* __restrict__ fc1T, const float* __restrict__ fc1_b,
    const unsigned short* __restrict__ fcmT, const float* __restrict__ fcm_b,
    const unsigned short* __restrict__ fc2T, const float* __restrict__ fc2_b,
    const float* __restrict__ meanA, const float* __restrict__ rstdA,
    float* __restrict__ zp_f32, unsigned short* __restrict__ zp_bf) {
    __shared__ unsigned char smem[65536];
    unsigned short* A0   = (unsigned short*)(smem);
    float*          xnb  = (float*)(smem + 8192);
    float*          zbuf = (float*)(smem + 32768);
    unsigned short* h1   = (unsigned short*)(smem + 32768);
    unsigned short* h2   = (unsigned short*)(smem);

    const int blk = blockIdx.x;
    const int bv = blk >> 1, dh = blk & 1;
    const int b = bv >> 6, v = bv & 63;
    const int t = threadIdx.x, lane = t & 31, wv = t >> 5;

    // Phase A: RevIN-normalized series into LDS
    const float mu = meanA[bv], rsd = rstdA[bv];
    for (int i = t; i < 1024; i += 256)
        xnb[i] = (x[((size_t)b * 1024 + i) * 64 + v] - mu) * rsd;
    __syncthreads();

    // Phase B: encode, z[ph][d] = patch . enc_w + enc_b (full D for LN stats)
    for (int idx = t; idx < 8192; idx += 256) {
        int ph = idx >> 7, d = idx & 127;
        float s = enc_b[d];
        const float* xp = xnb + ph * 16;
#pragma unroll
        for (int pl = 0; pl < 16; ++pl) s += xp[pl] * enc_w[pl * 128 + d];
        zbuf[idx] = s;
    }
    __syncthreads();

    // Phase B2: LayerNorm each ph-row over D=128; keep our d-half transposed
    for (int ph = wv; ph < 64; ph += 8) {
        const float* row = zbuf + ph * 128;
        float vals[4]; float s = 0.f;
#pragma unroll
        for (int j = 0; j < 4; ++j) { vals[j] = row[lane * 4 + j]; s += vals[j]; }
        s = wave_sum(s);
        float m = s * (1.f / 128.f);
        float q = 0.f;
#pragma unroll
        for (int j = 0; j < 4; ++j) { float dd = vals[j] - m; q += dd * dd; }
        q = wave_sum(q);
        float rr = rsqrtf(q * (1.f / 128.f) + 1e-5f);
#pragma unroll
        for (int j = 0; j < 4; ++j) {
            int d = lane * 4 + j;
            int dl = d - dh * 64;
            if (dl >= 0 && dl < 64) {
                float o = (vals[j] - m) * rr * ln_w[d] + ln_b[d];
                A0[dl * 64 + ph] = f2bf(o);
            }
        }
    }
    __syncthreads();

    // Phase C: fc1  [64 x 64] @ [64 x 256] -> h1, +bias, relu
    for (int tt = wv; tt < 64; tt += 8) {
        int tm = tt >> 4, tn = tt & 15;
        floatx8 acc = zero8();
#pragma unroll
        for (int kb = 0; kb < 64; kb += 32) {
            bf16x16 a  = load_frag_a(A0 + (tm * 16) * 64 + kb, 64, lane);
            bf16x16 bb = load_frag_b(fc1T + (tn * 16) * 64 + kb, 64, lane);
            acc = wmma_bf16(a, bb, acc);
        }
        int n = lane & 15, g = lane >> 4;
        int col = tn * 16 + n;
        float bias = fc1_b[col];
#pragma unroll
        for (int e = 0; e < 8; ++e) {
            int m = tm * 16 + g * 8 + e;
            float val = acc[e] + bias;
            h1[m * 256 + col] = f2bf(val > 0.f ? val : 0.f);
        }
    }
    __syncthreads();

    // Phase D: fcm in two 256-col halves; fc2 accumulated in VGPRs across halves
    const int tm2 = wv >> 1, tn2 = wv & 1;
    floatx8 cacc = zero8();
    for (int hh = 0; hh < 2; ++hh) {
        for (int tt = wv; tt < 64; tt += 8) {
            int tm = tt >> 4, tn = tt & 15;
            floatx8 acc = zero8();
            for (int kb = 0; kb < 256; kb += 32) {
                bf16x16 a  = load_frag_a(h1 + (tm * 16) * 256 + kb, 256, lane);
                bf16x16 bb = load_frag_b(fcmT + (hh * 256 + tn * 16) * 256 + kb, 256, lane);
                acc = wmma_bf16(a, bb, acc);
            }
            int n = lane & 15, g = lane >> 4;
            int colg = hh * 256 + tn * 16 + n;
            int coll = tn * 16 + n;
            float bias = fcm_b[colg];
#pragma unroll
            for (int e = 0; e < 8; ++e) {
                int m = tm * 16 + g * 8 + e;
                float val = acc[e] + bias;
                h2[m * 256 + coll] = f2bf(val > 0.f ? val : 0.f);
            }
        }
        __syncthreads();
        for (int kb = 0; kb < 256; kb += 32) {
            bf16x16 a  = load_frag_a(h2 + (tm2 * 16) * 256 + kb, 256, lane);
            bf16x16 bb = load_frag_b(fc2T + (tn2 * 16) * 512 + hh * 256 + kb, 512, lane);
            cacc = wmma_bf16(a, bb, cacc);
        }
        __syncthreads();
    }

    // Phase E: z_p transposed out: [bv][pf][d], fp32 + bf16
    {
        int n = lane & 15, g = lane >> 4;
#pragma unroll
        for (int e = 0; e < 8; ++e) {
            int d = dh * 64 + tm2 * 16 + g * 8 + e;
            int pf = tn2 * 16 + n;
            float val = cacc[e] + fc2_b[pf];
            size_t off = ((size_t)bv * 32 + pf) * 128 + d;
            zp_f32[off] = val;
            zp_bf[off]  = f2bf(val);
        }
    }
}

// ---------------- soft-VQ GEMM -> logits ----------------
// one wave per 16x16 tile of [N=65536 x K=1024]; logits = 2*z.c - |z|^2 - |c|^2
__global__ __launch_bounds__(256) void k_vq(
    const unsigned short* __restrict__ zbf, const unsigned short* __restrict__ centbf,
    const float* __restrict__ z2, const float* __restrict__ c2,
    float* __restrict__ logits) {
    int gw = blockIdx.x * 8 + (threadIdx.x >> 5);
    int lane = threadIdx.x & 31;
    int tN = gw & 63, tM = gw >> 6;
    floatx8 acc = zero8();
#pragma unroll
    for (int kb = 0; kb < 128; kb += 32) {
        bf16x16 a  = load_frag_a(zbf + ((size_t)tM * 16) * 128 + kb, 128, lane);
        bf16x16 bb = load_frag_b(centbf + (tN * 16) * 128 + kb, 128, lane);
        acc = wmma_bf16(a, bb, acc);
    }
    int n = lane & 15, g = lane >> 4;
    int col = tN * 16 + n;
    float cc = c2[col];
#pragma unroll
    for (int e = 0; e < 8; ++e) {
        int row = tM * 16 + g * 8 + e;
        logits[(size_t)row * 1024 + col] = 2.f * acc[e] - z2[row] - cc;
    }
}

// ---------------- top-5 + softmax + codebook gather ----------------
__global__ __launch_bounds__(256) void k_topk(
    const float* __restrict__ logits, const float* __restrict__ centroids,
    float* __restrict__ zcode) {
    int n = blockIdx.x * 8 + (threadIdx.x >> 5);
    int lane = threadIdx.x & 31;
    const float* lrow = logits + (size_t)n * 1024;
    float rv[32];
#pragma unroll
    for (int j = 0; j < 32; ++j) rv[j] = lrow[j * 32 + lane];
    float w5[5]; int id5[5];
    for (int p = 0; p < 5; ++p) {
        float bvv = -3.4e38f; int bk = 1 << 30;
#pragma unroll
        for (int j = 0; j < 32; ++j) {
            int kidx = j * 32 + lane;
            if (rv[j] > bvv) { bvv = rv[j]; bk = kidx; }
        }
        for (int o = 16; o; o >>= 1) {
            float ov = __shfl_xor(bvv, o, 32);
            int ok = __shfl_xor(bk, o, 32);
            if (ov > bvv || (ov == bvv && ok < bk)) { bvv = ov; bk = ok; }
        }
        w5[p] = bvv; id5[p] = bk;
        if ((bk & 31) == lane) rv[bk >> 5] = -3.4e38f;
    }
    float mx = w5[0];  // sorted descending
    float e5[5]; float sum = 0.f;
#pragma unroll
    for (int p = 0; p < 5; ++p) { e5[p] = __expf(w5[p] - mx); sum += e5[p]; }
    float inv = 1.f / sum;
#pragma unroll
    for (int j = 0; j < 4; ++j) {
        int d = lane * 4 + j;
        float s = 0.f;
#pragma unroll
        for (int p = 0; p < 5; ++p) s += e5[p] * centroids[(size_t)id5[p] * 128 + d];
        zcode[(size_t)n * 128 + d] = s * inv;
    }
}

// ---------------- fuse GEMM + LN + decode + denorm ----------------
// one block per 16 rows of N; wave w computes cols [16w,16w+16)
__global__ __launch_bounds__(256) void k_fuse(
    const unsigned short* __restrict__ zp_bf, const unsigned short* __restrict__ fuseT,
    const float* __restrict__ fuse_b, const float* __restrict__ zcode,
    const float* __restrict__ fln_w, const float* __restrict__ fln_b,
    const float* __restrict__ dec_w, const float* __restrict__ dec_b,
    const float* __restrict__ meanA, const float* __restrict__ stdA,
    float* __restrict__ outp) {
    __shared__ float fs[16 * 128];
    int rb = blockIdx.x;
    int t = threadIdx.x, lane = t & 31, wv = t >> 5;
    floatx8 acc = zero8();
#pragma unroll
    for (int kb = 0; kb < 128; kb += 32) {
        bf16x16 a  = load_frag_a(zp_bf + ((size_t)rb * 16) * 128 + kb, 128, lane);
        bf16x16 bb = load_frag_b(fuseT + (wv * 16) * 128 + kb, 128, lane);
        acc = wmma_bf16(a, bb, acc);
    }
    int nn = lane & 15, g = lane >> 4;
    int col = wv * 16 + nn;
    float bias = fuse_b[col];
#pragma unroll
    for (int e = 0; e < 8; ++e) {
        int r = g * 8 + e;
        float val = acc[e] + bias;
        val = val > 0.f ? val : 0.f;
        fs[r * 128 + col] = val + zcode[((size_t)rb * 16 + r) * 128 + col];
    }
    __syncthreads();
    // LayerNorm per row (2 rows per wave)
    for (int rr = 0; rr < 2; ++rr) {
        int r = wv * 2 + rr;
        float* row = fs + r * 128;
        float vals[4]; float s = 0.f;
#pragma unroll
        for (int j = 0; j < 4; ++j) { vals[j] = row[lane * 4 + j]; s += vals[j]; }
        s = wave_sum(s);
        float m = s * (1.f / 128.f);
        float q = 0.f;
#pragma unroll
        for (int j = 0; j < 4; ++j) { float dd = vals[j] - m; q += dd * dd; }
        q = wave_sum(q);
        float rr2 = rsqrtf(q * (1.f / 128.f) + 1e-5f);
#pragma unroll
        for (int j = 0; j < 4; ++j) {
            int d = lane * 4 + j;
            row[d] = (vals[j] - m) * rr2 * fln_w[d] + fln_b[d];
        }
    }
    __syncthreads();
    // decode D->16 and RevIN denorm; thread = (row, pl)
    int r = t >> 4, pl = t & 15;
    float s = dec_b[pl];
    for (int d = 0; d < 128; ++d) s += fs[r * 128 + d] * dec_w[d * 16 + pl];
    int n = rb * 16 + r;
    int b = n >> 11, v = (n >> 5) & 63, pf = n & 31;
    int pred = pf * 16 + pl;
    outp[((size_t)b * 512 + pred) * 64 + v] = s * stdA[b * 64 + v] + meanA[b * 64 + v];
}

extern "C" void kernel_launch(void* const* d_in, const int* in_sizes, int n_in,
                              void* d_out, int out_size, void* d_ws, size_t ws_size,
                              hipStream_t stream) {
    (void)in_sizes; (void)n_in; (void)out_size; (void)ws_size;
    const float* x      = (const float*)d_in[0];
    const float* cent   = (const float*)d_in[1];
    const float* enc_w  = (const float*)d_in[2];
    const float* enc_b  = (const float*)d_in[3];
    const float* ln_w   = (const float*)d_in[4];
    const float* ln_b   = (const float*)d_in[5];
    const float* fc1_w  = (const float*)d_in[6];
    const float* fc1_b  = (const float*)d_in[7];
    const float* fcm_w  = (const float*)d_in[8];
    const float* fcm_b  = (const float*)d_in[9];
    const float* fc2_w  = (const float*)d_in[10];
    const float* fc2_b  = (const float*)d_in[11];
    const float* fuse_w = (const float*)d_in[12];
    const float* fuse_b = (const float*)d_in[13];
    const float* fln_w  = (const float*)d_in[14];
    const float* fln_b  = (const float*)d_in[15];
    const float* dec_w  = (const float*)d_in[16];
    const float* dec_b  = (const float*)d_in[17];

    unsigned char* ws = (unsigned char*)d_ws;
    float* meanA = (float*)(ws + 0);          // 2048 f
    float* stdA  = (float*)(ws + 8192);       // 2048 f
    float* rstdA = (float*)(ws + 16384);      // 2048 f
    float* c2    = (float*)(ws + 24576);      // 1024 f
    float* z2    = (float*)(ws + 28672);      // 65536 f
    unsigned short* fc1T   = (unsigned short*)(ws + 290816);   // [256][64]
    unsigned short* fcmT   = (unsigned short*)(ws + 323584);   // [512][256]
    unsigned short* fc2T   = (unsigned short*)(ws + 585728);   // [32][512]
    unsigned short* fuseT  = (unsigned short*)(ws + 618496);   // [128][128]
    unsigned short* centbf = (unsigned short*)(ws + 651264);   // [1024][128]
    float*          zp_f32 = (float*)(ws + 913408);            // [65536][128]
    unsigned short* zp_bf  = (unsigned short*)(ws + 913408 + 33554432);
    float*          zcode  = (float*)(ws + 913408 + 33554432 + 16777216);

    float* outp   = (float*)d_out;
    float* logits = outp + 32 * 512 * 64;  // [65536][1024]

    k_transpose_bf16<<<64,  256, 0, stream>>>(fc1_w, fc1T, 64, 256);
    k_transpose_bf16<<<512, 256, 0, stream>>>(fcm_w, fcmT, 256, 512);
    k_transpose_bf16<<<64,  256, 0, stream>>>(fc2_w, fc2T, 512, 32);
    k_transpose_bf16<<<64,  256, 0, stream>>>(fuse_w, fuseT, 128, 128);
    k_convert_bf16<<<512, 256, 0, stream>>>(cent, centbf, 1024 * 128);
    k_rowsq<<<128, 256, 0, stream>>>(cent, c2, 1024);
    k_revin<<<2048, 256, 0, stream>>>(x, meanA, stdA, rstdA);
    k_mlp<<<4096, 256, 0, stream>>>(x, enc_w, enc_b, ln_w, ln_b, fc1T, fc1_b,
                                    fcmT, fcm_b, fc2T, fc2_b, meanA, rstdA,
                                    zp_f32, zp_bf);
    k_rowsq<<<8192, 256, 0, stream>>>(zp_f32, z2, 65536);
    k_vq<<<32768, 256, 0, stream>>>(zp_bf, centbf, z2, c2, logits);
    k_topk<<<8192, 256, 0, stream>>>(logits, cent, zcode);
    k_fuse<<<4096, 256, 0, stream>>>(zp_bf, fuseT, fuse_b, zcode, fln_w, fln_b,
                                     dec_w, dec_b, meanA, stdA, outp);
}
